// GATNet_7584912244762
// MI455X (gfx1250) — compile-verified, matrix operand order checked by reading, not run on hardware
//
#include <hip/hip_runtime.h>

#define NEG_SLOPE 0.2f
#define GAT_EPS 1e-16f

typedef __attribute__((ext_vector_type(16))) __bf16 v16bf;
typedef __attribute__((ext_vector_type(8)))  float  v8f;
typedef __attribute__((ext_vector_type(4)))  unsigned u32x4;
typedef __attribute__((ext_vector_type(8)))  int      i32x8;
typedef __attribute__((ext_vector_type(4)))  int      i32x4;

// pack two f32 into one dword of {lo=bf16(a), hi=bf16(b)} with a single cvt
__device__ __forceinline__ unsigned pk2(float a, float b) {
  unsigned r;
  asm("v_cvt_pk_bf16_f32 %0, %1, %2" : "=v"(r) : "v"(a), "v"(b));
  return r;
}

// ---------------------------------------------------------------------------
// GEMM: C[M,Nc] = A[M,K] @ W[K,Nc]  (f32 in/out, bf16 WMMA, f32 accumulate)
// block = 256 threads = 8 wave32 waves; block tile 128(M) x 16(N); k-step 32.
//   A tile (128x32 f32) fetched by the Tensor Data Mover (tensor_load_to_lds,
//     TENSORcnt), OOB rows of the last block zero-filled by the D# dims.
//   B tile (32x16 f32) fetched with global_load_async_to_lds_b128 (ASYNCcnt).
//   Both repacked into bf16-pair layouts so each WMMA fragment is 2x
//   contiguous ds_load_b128 per the ISA 16-bit operand layouts.
// ---------------------------------------------------------------------------
__global__ __launch_bounds__(256)
void gemm_bf16_wmma(const float* __restrict__ A, const float* __restrict__ W,
                    float* __restrict__ C, int M, int K, int Nc) {
  __shared__ float    Araw[128 * 32];  // TDM-staged raw f32 A tile   (16 KB)
  __shared__ unsigned AsU[128 * 16];   // [row][kpair] packed bf16    ( 8 KB)
  __shared__ float    Braw[32 * 16];   // async-staged raw f32 B tile ( 2 KB)
  __shared__ unsigned BsU[16 * 16];    // [n][kpair]  packed bf16     ( 1 KB)

  const int tid   = threadIdx.x;
  const int lane  = tid & 31;
  const int wave  = tid >> 5;
  const int mBlk  = blockIdx.x * 128;
  const int nTile = blockIdx.y * 16;
  const int mWave = mBlk + wave * 16;
  const int mRow  = lane & 15;      // M index for A frag, N index for B/D frags
  const int half  = lane >> 4;
  const unsigned arawLds = (unsigned)(size_t)(void*)Araw;  // LDS offset = low 32b
  const unsigned brawLds = (unsigned)(size_t)(void*)Braw;

  int rows = M - mBlk; if (rows > 128) rows = 128;   // partial last block

  v8f acc = {};
  for (int k0 = 0; k0 < K; k0 += 32) {
    // ---- TDM: DMA the 128x32 f32 A tile into LDS (one issue per block) ---
    if (wave == 0) {
      unsigned long long ga =
          (unsigned long long)(size_t)(A + (size_t)mBlk * K + k0);
      u32x4 g0 = { 1u,                                   // count=1, user mode
                   arawLds,                              // lds_addr
                   (unsigned)(ga & 0xffffffffu),         // global_addr[31:0]
                   (unsigned)((ga >> 32) & 0x01ffffffu)  // global_addr[56:32]
                     | 0x80000000u };                    // type=2 (image)
      i32x8 g1 = { (int)0x00020000u,                     // data_size=4B
                   (int)0x00200000u,                     // tensor_dim0=32 (lo16<<16)
                   (int)((unsigned)rows << 16),          // tensor_dim1=rows (lo16<<16)
                   (int)0x00200000u,                     // tile_dim0=32
                   128,                                  // tile_dim1=128
                   K,                                    // tensor_dim0_stride=K
                   0, 0 };
      i32x4 z4 = { 0, 0, 0, 0 };
      i32x8 z8 = { 0, 0, 0, 0, 0, 0, 0, 0 };
      __builtin_amdgcn_tensor_load_to_lds(g0, g1, z4, z4, z8, 0);
      __builtin_amdgcn_s_wait_tensorcnt(0);
    }
    // ---- async-stage raw B tile (32x16 f32) straight into LDS ------------
    if (tid < 128) {
      int row  = tid >> 2;
      int col4 = (tid & 3) * 4;
      const float* gsrc = W + (size_t)(k0 + row) * Nc + nTile + col4;
      unsigned dst = brawLds + (unsigned)tid * 16u;
      asm volatile("global_load_async_to_lds_b128 %0, %1, off"
                   :: "v"(dst), "v"(gsrc) : "memory");
    }
    asm volatile("s_wait_asynccnt 0x0" ::: "memory");
    __syncthreads();

    // ---- repack A: raw f32 -> [row][kpair] packed bf16 (2048 dwords) -----
    #pragma unroll
    for (int j = 0; j < 8; ++j) {
      int u   = j * 256 + tid;
      int row = u >> 4;
      int kp  = u & 15;
      float2 f = *(const float2*)(Araw + row * 32 + 2 * kp);
      AsU[u] = pk2(f.x, f.y);
    }
    // ---- repack B: raw [k][n] f32 -> [n][kpair] packed bf16 --------------
    {
      int n  = tid & 15;
      int kp = tid >> 4;                 // 0..15
      BsU[n * 16 + kp] = pk2(Braw[(2 * kp) * 16 + n], Braw[(2 * kp + 1) * 16 + n]);
    }
    __syncthreads();

    // ---- build fragments: 2x ds_load_b128 each ---------------------------
    union Frag { v16bf v; uint4 q[2]; };
    Frag af, bf;
    const unsigned arow = (unsigned)(wave * 16 + mRow) * 16u;
    af.q[0] = *(const uint4*)(AsU + arow + 4 * half);
    af.q[1] = *(const uint4*)(AsU + arow + 8 + 4 * half);
    bf.q[0] = *(const uint4*)(BsU + mRow * 16 + 4 * half);
    bf.q[1] = *(const uint4*)(BsU + mRow * 16 + 8 + 4 * half);

    acc = __builtin_amdgcn_wmma_f32_16x16x32_bf16(false, af.v, false, bf.v,
                                                  (short)0, acc, false, false);
    __syncthreads();
  }
  // store per 16x16 f32 C/D layout: VGPR r -> M = 8*half + r, N = lane%16
  #pragma unroll
  for (int r = 0; r < 8; ++r) {
    int row = mWave + half * 8 + r;
    if (row < M) C[(size_t)row * Nc + nTile + mRow] = acc[r];
  }
}

// ---------------------------------------------------------------------------
// order-preserving float <-> uint encoding for atomic max on floats
// ---------------------------------------------------------------------------
__device__ __forceinline__ unsigned fenc(float f) {
  unsigned u = __float_as_uint(f);
  return (u & 0x80000000u) ? ~u : (u | 0x80000000u);
}
__device__ __forceinline__ float fdec(unsigned k) {
  unsigned u = (k & 0x80000000u) ? (k & 0x7fffffffu) : ~k;
  return __uint_as_float(u);
}

__global__ void fill_f32(float* p, float v, long n) {
  long i = (long)blockIdx.x * blockDim.x + threadIdx.x;
  if (i < n) p[i] = v;
}
__global__ void fill_u32(unsigned* p, unsigned v, long n) {
  long i = (long)blockIdx.x * blockDim.x + threadIdx.x;
  if (i < n) p[i] = v;
}

// alpha_src / alpha_dst node scores: thread per (node, head), C-length dot
__global__ void attn_score(const float* __restrict__ h,
                           const float* __restrict__ a_src,
                           const float* __restrict__ a_dst,
                           float* __restrict__ as, float* __restrict__ ad,
                           int N, int H, int C) {
  int i = blockIdx.x * blockDim.x + threadIdx.x;
  if (i >= N * H) return;
  int n = i / H, hh = i % H;
  const float* row = h + (size_t)n * H * C + (size_t)hh * C;
  const float* ws = a_src + hh * C;
  const float* wd = a_dst + hh * C;
  float s = 0.f, d = 0.f;
  #pragma unroll 4
  for (int c = 0; c < C; ++c) { float v = row[c]; s += v * ws[c]; d += v * wd[c]; }
  as[i] = s; ad[i] = d;
}

__device__ __forceinline__ void edge_nodes(const int* ei, int e, int E, int* s, int* d) {
  if (e < E) { *s = ei[e]; *d = ei[E + e]; }
  else       { *s = *d = e - E; }  // self loop
}

// pass 1: segment max of leaky_relu scores over dst
__global__ void edge_max(const int* __restrict__ ei, const float* __restrict__ as,
                         const float* __restrict__ ad, unsigned* __restrict__ mx,
                         int E, int N, int H) {
  long idx = (long)blockIdx.x * blockDim.x + threadIdx.x;
  long tot = (long)(E + N) * H;
  if (idx >= tot) return;
  int e = (int)(idx / H), hh = (int)(idx % H);
  int s, d; edge_nodes(ei, e, E, &s, &d);
  float v = as[(size_t)s * H + hh] + ad[(size_t)d * H + hh];
  v = v > 0.f ? v : NEG_SLOPE * v;
  atomicMax(&mx[(size_t)d * H + hh], fenc(v));
}

// pass 2: exp(e - max) cached per edge, segment-sum denominator over dst
__global__ void edge_exp(const int* __restrict__ ei, const float* __restrict__ as,
                         const float* __restrict__ ad, const unsigned* __restrict__ mx,
                         float* __restrict__ eexp, float* __restrict__ den,
                         int E, int N, int H) {
  long idx = (long)blockIdx.x * blockDim.x + threadIdx.x;
  long tot = (long)(E + N) * H;
  if (idx >= tot) return;
  int e = (int)(idx / H), hh = (int)(idx % H);
  int s, d; edge_nodes(ei, e, E, &s, &d);
  float v = as[(size_t)s * H + hh] + ad[(size_t)d * H + hh];
  v = v > 0.f ? v : NEG_SLOPE * v;
  float t = __expf(v - fdec(mx[(size_t)d * H + hh]));
  eexp[idx] = t;
  atomicAdd(&den[(size_t)d * H + hh], t);
}

// pass 3: out[dst,h,:] += h[src,h,:] * (eexp / (den+eps)); thread per (edge, c)
__global__ void edge_aggr(const int* __restrict__ ei, const float* __restrict__ h,
                          const float* __restrict__ eexp, const float* __restrict__ den,
                          float* __restrict__ out, int E, int N, int H, int C) {
  long idx = (long)blockIdx.x * blockDim.x + threadIdx.x;
  long tot = (long)(E + N) * C;
  if (idx >= tot) return;
  int e = (int)(idx / C), c = (int)(idx % C);
  int s, d; edge_nodes(ei, e, E, &s, &d);
  for (int hh = 0; hh < H; ++hh) {
    float alpha = eexp[(size_t)e * H + hh] / (den[(size_t)d * H + hh] + GAT_EPS);
    atomicAdd(&out[((size_t)d * H + hh) * C + c],
              h[((size_t)s * H + hh) * C + c] * alpha);
  }
}

// x = elu(x + bias[i % F])
__global__ void elu_bias(float* __restrict__ x, const float* __restrict__ b,
                         long n, int F) {
  long i = (long)blockIdx.x * blockDim.x + threadIdx.x;
  if (i >= n) return;
  float v = x[i] + b[i % F];
  x[i] = v > 0.f ? v : (__expf(v) - 1.f);
}

// per-group sum + counts
__global__ void pool_sum(const float* __restrict__ h, const int* __restrict__ batch,
                         float* __restrict__ pooled, float* __restrict__ cnt,
                         int N, int C) {
  long i = (long)blockIdx.x * blockDim.x + threadIdx.x;
  if (i >= (long)N * C) return;
  int n = (int)(i / C), c = (int)(i % C);
  int g = batch[n];
  atomicAdd(&pooled[(size_t)g * C + c], h[i]);
  if (c == 0) atomicAdd(&cnt[g], 1.f);
}

// out[g] = (pooled[g]/max(cnt,1)) . Wc + bc
__global__ void classify(const float* __restrict__ pooled, const float* __restrict__ cnt,
                         const float* __restrict__ Wc, const float* __restrict__ bc,
                         float* __restrict__ out, int G, int C) {
  int g = blockIdx.x * blockDim.x + threadIdx.x;
  if (g >= G) return;
  float inv = 1.f / fmaxf(cnt[g], 1.f);
  float s = 0.f;
  for (int c = 0; c < C; ++c) s += pooled[(size_t)g * C + c] * inv * Wc[c];
  out[g] = s + bc[0];
}

// ---------------------------------------------------------------------------
extern "C" void kernel_launch(void* const* d_in, const int* in_sizes, int n_in,
                              void* d_out, int out_size, void* d_ws, size_t ws_size,
                              hipStream_t stream) {
  const float* x     = (const float*)d_in[0];
  const int*   ei    = (const int*)d_in[1];
  const int*   batch = (const int*)d_in[2];
  const float* W1    = (const float*)d_in[3];
  const float* asr1  = (const float*)d_in[4];
  const float* adt1  = (const float*)d_in[5];
  const float* b1    = (const float*)d_in[6];
  const float* W2    = (const float*)d_in[7];
  const float* asr2  = (const float*)d_in[8];
  const float* adt2  = (const float*)d_in[9];
  const float* b2    = (const float*)d_in[10];
  const float* Wc    = (const float*)d_in[11];
  const float* bc    = (const float*)d_in[12];

  const int IN = 128, HID = 64, HEADS = 4, G = 64;
  const int N = in_sizes[0] / IN;
  const int E = in_sizes[1] / 2;
  const int F1 = HEADS * HID;           // 256
  const long Etot = (long)E + N;

  // ---- workspace carve-out (layer 2 aliases layer 1 regions) ----
  char* p = (char*)d_ws;
  auto carve = [&](size_t bytes) {
    void* r = (void*)p;
    p += (bytes + 255) & ~(size_t)255;
    return r;
  };
  float*    h1    = (float*)carve((size_t)N * F1 * 4);   // x@W1        (51.2MB)
  float*    o1    = (float*)carve((size_t)N * F1 * 4);   // conv1 out   (51.2MB)
  float*    as1   = (float*)carve((size_t)N * HEADS * 4);
  float*    ad1   = (float*)carve((size_t)N * HEADS * 4);
  unsigned* mx1   = (unsigned*)carve((size_t)N * HEADS * 4);
  float*    den1  = (float*)carve((size_t)N * HEADS * 4);
  float*    eexp1 = (float*)carve((size_t)Etot * HEADS * 4);
  float*    pooled= (float*)carve((size_t)G * HID * 4);
  float*    cnt   = (float*)carve((size_t)G * 4);
  // layer-2 aliases (layer-1 buffers dead by then)
  float*    h2    = h1;                    // [N,64]
  float*    o2    = h1 + (size_t)N * HID;  // [N,64], disjoint from h2
  float*    as2   = as1;
  float*    ad2   = ad1;
  unsigned* mx2   = mx1;
  float*    den2  = den1;
  float*    eexp2 = eexp1;

  const int TB = 256;
  auto blocks = [&](long n) { return (unsigned)((n + TB - 1) / TB); };

  // ================= conv1: GATConv(128 -> 64, heads=4, concat) ============
  {
    dim3 g((N + 127) / 128, F1 / 16);
    gemm_bf16_wmma<<<g, TB, 0, stream>>>(x, W1, h1, N, IN, F1);
  }
  attn_score<<<blocks((long)N * HEADS), TB, 0, stream>>>(h1, asr1, adt1, as1, ad1, N, HEADS, HID);
  fill_u32<<<blocks((long)N * HEADS), TB, 0, stream>>>(mx1, 0u, (long)N * HEADS);
  fill_f32<<<blocks((long)N * HEADS), TB, 0, stream>>>(den1, 0.f, (long)N * HEADS);
  fill_f32<<<blocks((long)N * F1), TB, 0, stream>>>(o1, 0.f, (long)N * F1);
  edge_max <<<blocks(Etot * HEADS), TB, 0, stream>>>(ei, as1, ad1, mx1, E, N, HEADS);
  edge_exp <<<blocks(Etot * HEADS), TB, 0, stream>>>(ei, as1, ad1, mx1, eexp1, den1, E, N, HEADS);
  edge_aggr<<<blocks(Etot * HID),   TB, 0, stream>>>(ei, h1, eexp1, den1, o1, E, N, HEADS, HID);
  elu_bias <<<blocks((long)N * F1), TB, 0, stream>>>(o1, b1, (long)N * F1, F1);

  // ================= conv2: GATConv(256 -> 64, heads=1) ====================
  {
    dim3 g((N + 127) / 128, HID / 16);
    gemm_bf16_wmma<<<g, TB, 0, stream>>>(o1, W2, h2, N, F1, HID);
  }
  attn_score<<<blocks((long)N), TB, 0, stream>>>(h2, asr2, adt2, as2, ad2, N, 1, HID);
  fill_u32<<<blocks((long)N), TB, 0, stream>>>(mx2, 0u, (long)N);
  fill_f32<<<blocks((long)N), TB, 0, stream>>>(den2, 0.f, (long)N);
  fill_f32<<<blocks((long)N * HID), TB, 0, stream>>>(o2, 0.f, (long)N * HID);
  edge_max <<<blocks(Etot), TB, 0, stream>>>(ei, as2, ad2, mx2, E, N, 1);
  edge_exp <<<blocks(Etot), TB, 0, stream>>>(ei, as2, ad2, mx2, eexp2, den2, E, N, 1);
  edge_aggr<<<blocks(Etot * HID), TB, 0, stream>>>(ei, h2, eexp2, den2, o2, E, N, 1, HID);
  elu_bias <<<blocks((long)N * HID), TB, 0, stream>>>(o2, b2, (long)N * HID, HID);

  // ================= mean pool per graph + classifier ======================
  fill_f32<<<blocks((long)G * HID), TB, 0, stream>>>(pooled, 0.f, (long)G * HID);
  fill_f32<<<1, TB, 0, stream>>>(cnt, 0.f, (long)G);
  pool_sum<<<blocks((long)N * HID), TB, 0, stream>>>(o2, batch, pooled, cnt, N, HID);
  classify<<<1, G, 0, stream>>>(pooled, cnt, Wc, bc, (float*)d_out, G, HID);
}
